// LSTM_MNIST_2layers_11708080849332
// MI455X (gfx1250) — compile-verified, matrix operand order checked by reading, not run on hardware
//
#include <hip/hip_runtime.h>
#include <hip/hip_bf16.h>

// ---------------------------------------------------------------------------
// 2-layer LSTM (I=128, H=512, B=256, T=256) -> linear head (C=10)
// Recurrent GEMMs run as bf16 WMMA (v_wmma_f32_16x16x32_bf16), f32 accumulate.
// Weights are converted once to bf16 and stored K-contiguous per output column
// so WMMA A/B fragments are plain 2x b128 loads per lane (CDNA5 layout).
// ---------------------------------------------------------------------------

typedef __bf16 v16bf  __attribute__((ext_vector_type(16)));
typedef __bf16 bf16x8 __attribute__((ext_vector_type(8)));
typedef float  v8f    __attribute__((ext_vector_type(8)));

#define I_   128
#define H_   512
#define C_   10
#define B_   256
#define T_   256
#define N4H  2048            // 4*H
#define K1   640             // I + H   (layer-1 fused K)
#define K2   1024            // 2*H     (layer-2 fused K)

// round-to-nearest-even f32 -> bf16 bits
__device__ __forceinline__ unsigned short f2bf(float x) {
    unsigned u = __float_as_uint(x);
    unsigned r = u + 0x7fffu + ((u >> 16) & 1u);
    return (unsigned short)(r >> 16);
}

__device__ __forceinline__ float sigm(float x) {
    return 1.0f / (1.0f + __expf(-x));
}

// ---------------------------------------------------------------------------
// Prep: build W1T [2048][640] and W2T [2048][1024] (bf16, K-contiguous rows).
// Column n = g*512 + h.  W1 rows: k<128 -> Wx1[g][k][h], else Wh1[g][k-128][h].
//                        W2 rows: k<512 -> Wx2[g][k][h], else Wh2[g][k-512][h].
// ---------------------------------------------------------------------------
__global__ __launch_bounds__(256)
void prep_weights(const float* __restrict__ Wx1, const float* __restrict__ Wh1,
                  const float* __restrict__ Wx2, const float* __restrict__ Wh2,
                  unsigned short* __restrict__ W1T, unsigned short* __restrict__ W2T)
{
    const int TOT1 = N4H * K1;                    // 1,310,720
    const int TOT2 = N4H * K2;                    // 2,097,152
    int idx = blockIdx.x * 256 + threadIdx.x;
    if (idx < TOT1) {
        int n = idx / K1, k = idx % K1;
        int g = n >> 9, h = n & 511;
        float v = (k < I_) ? Wx1[(g * I_ + k) * H_ + h]
                           : Wh1[(g * H_ + (k - I_)) * H_ + h];
        W1T[idx] = f2bf(v);
    } else if (idx < TOT1 + TOT2) {
        int j = idx - TOT1;
        int n = j / K2, k = j % K2;
        int g = n >> 9, h = n & 511;
        float v = (k < H_) ? Wx2[(g * H_ + k) * H_ + h]
                           : Wh2[(g * H_ + (k - H_)) * H_ + h];
        W2T[j] = f2bf(v);
    }
}

__global__ __launch_bounds__(256)
void convert_x(const float* __restrict__ x, unsigned short* __restrict__ xbf, int n)
{
    int idx = blockIdx.x * 256 + threadIdx.x;
    if (idx < n) xbf[idx] = f2bf(x[idx]);
}

__global__ __launch_bounds__(256)
void zero_state(unsigned short* h1b, unsigned short* h2b,
                float* c1, float* c2, float* h1f, float* h2f)
{
    int idx = blockIdx.x * 256 + threadIdx.x;   // 0 .. B*H-1
    h1b[idx] = 0; h2b[idx] = 0;
    c1[idx] = 0.f; c2[idx] = 0.f; h1f[idx] = 0.f; h2f[idx] = 0.f;
}

// ---------------------------------------------------------------------------
// WMMA GEMM:  Z[B, 2048] = concat(A0, A1 along K) @ WT^T
//   A0: rows of bf16, element stride sA0, covers K [0, K0)
//   A1: rows of bf16, element stride sA1, covers K [K0, Ktot)
//   WT: [2048][Ktot] bf16 (output-column-major, K contiguous)
// Block = 128x64 output tile, 8 waves, each wave 2x2 WMMA 16x16 tiles.
// Fragment layout per CDNA5 ISA: lane<16 -> K {0..7,16..23}, lane>=16 -> +8.
// ---------------------------------------------------------------------------
__global__ __launch_bounds__(256)
void lstm_gemm_wmma(const unsigned short* __restrict__ A0u, int sA0,
                    const unsigned short* __restrict__ A1u, int sA1,
                    int K0, int Ktot,
                    const unsigned short* __restrict__ WTu,
                    float* __restrict__ Z)
{
    const __bf16* A0 = (const __bf16*)A0u;
    const __bf16* A1 = (const __bf16*)A1u;
    const __bf16* WT = (const __bf16*)WTu;

    const int lane = threadIdx.x & 31;
    const int wv   = threadIdx.x >> 5;
    const int wm   = wv & 3;            // 0..3  (32-row strips)
    const int wn   = wv >> 2;           // 0..1  (32-col strips)
    const int row0 = blockIdx.x * 128 + wm * 32;
    const int col0 = blockIdx.y * 64  + wn * 32;
    const int lm     = lane & 15;
    const int hihalf = lane >> 4;       // 0|1
    const int kshift = hihalf ? 8 : 0;  // ISA K offset for upper half-wave

    v8f acc[2][2] = {};

    for (int kc = 0; kc < Ktot; kc += 32) {
        const __bf16* Abase;
        int stride, kloc;
        if (kc < K0) { Abase = A0; stride = sA0; kloc = kc; }
        else         { Abase = A1; stride = sA1; kloc = kc - K0; }

        v16bf a[2];
#pragma unroll
        for (int i = 0; i < 2; ++i) {
            const __bf16* p = Abase + (size_t)(row0 + i * 16 + lm) * stride + kloc + kshift;
            bf16x8 lo = *(const bf16x8*)(p);
            bf16x8 hi = *(const bf16x8*)(p + 16);
            a[i] = __builtin_shufflevector(lo, hi,
                     0,1,2,3,4,5,6,7,8,9,10,11,12,13,14,15);
        }

        v16bf b[2];
#pragma unroll
        for (int j = 0; j < 2; ++j) {
            const __bf16* p = WT + (size_t)(col0 + j * 16 + lm) * Ktot + kc + kshift;
            bf16x8 lo = *(const bf16x8*)(p);
            bf16x8 hi = *(const bf16x8*)(p + 16);
            b[j] = __builtin_shufflevector(lo, hi,
                     0,1,2,3,4,5,6,7,8,9,10,11,12,13,14,15);
            __builtin_prefetch(p + 32, 0, 0);   // next K chunk of weights
        }

#pragma unroll
        for (int i = 0; i < 2; ++i)
#pragma unroll
            for (int j = 0; j < 2; ++j)
                acc[i][j] = __builtin_amdgcn_wmma_f32_16x16x32_bf16(
                    false, a[i], false, b[j], (short)0, acc[i][j], false, false);
    }

    // C/D layout: VGPR r, lanes 0-15 -> M=r, lanes 16-31 -> M=8+r; N=lane&15.
#pragma unroll
    for (int i = 0; i < 2; ++i) {
        int rbase = row0 + i * 16 + hihalf * 8;
#pragma unroll
        for (int j = 0; j < 2; ++j) {
            int c = col0 + j * 16 + lm;
#pragma unroll
            for (int r = 0; r < 8; ++r)
                Z[(size_t)(rbase + r) * N4H + c] = acc[i][j][r];
        }
    }
}

// ---------------------------------------------------------------------------
// Elementwise LSTM cell update. Gate order [g, i, f, o] along N4H.
// ---------------------------------------------------------------------------
__global__ __launch_bounds__(256)
void lstm_update(const float* __restrict__ Z, const float* __restrict__ bias,
                 float* __restrict__ c, unsigned short* __restrict__ h_bf,
                 float* __restrict__ h_f32)
{
    int idx = blockIdx.x * 256 + threadIdx.x;   // 0 .. B*H-1
    int b = idx >> 9;
    int h = idx & 511;
    const float* zr = Z + (size_t)b * N4H;
    float zg = zr[0 * H_ + h] + bias[0 * H_ + h];
    float zi = zr[1 * H_ + h] + bias[1 * H_ + h];
    float zf = zr[2 * H_ + h] + bias[2 * H_ + h];
    float zo = zr[3 * H_ + h] + bias[3 * H_ + h];
    float g  = tanhf(zg);
    float ii = sigm(zi);
    float ff = sigm(zf);
    float oo = sigm(zo);
    float cn = g * ii + c[idx] * ff;
    float hn = tanhf(cn) * oo;
    c[idx]     = cn;
    h_f32[idx] = hn;
    h_bf[idx]  = f2bf(hn);
}

// ---------------------------------------------------------------------------
// Final projection: out[b][c] = h2[b] . Wph[:,c] + bp[c]   (tiny, f32)
// ---------------------------------------------------------------------------
__global__ __launch_bounds__(256)
void proj_kernel(const float* __restrict__ h2, const float* __restrict__ Wph,
                 const float* __restrict__ bp, float* __restrict__ out)
{
    int idx = blockIdx.x * 256 + threadIdx.x;   // 0 .. B*C-1
    if (idx >= B_ * C_) return;
    int b = idx / C_, cix = idx % C_;
    float s = bp[cix];
    const float* hr = h2 + (size_t)b * H_;
    for (int h = 0; h < H_; ++h) s += hr[h] * Wph[h * C_ + cix];
    out[idx] = s;
}

// ---------------------------------------------------------------------------
extern "C" void kernel_launch(void* const* d_in, const int* in_sizes, int n_in,
                              void* d_out, int out_size, void* d_ws, size_t ws_size,
                              hipStream_t stream)
{
    (void)in_sizes; (void)n_in; (void)out_size; (void)ws_size;

    const float* x   = (const float*)d_in[0];
    const float* Wx1 = (const float*)d_in[1];
    const float* Wh1 = (const float*)d_in[2];
    const float* b1  = (const float*)d_in[3];
    const float* Wx2 = (const float*)d_in[4];
    const float* Wh2 = (const float*)d_in[5];
    const float* b2  = (const float*)d_in[6];
    const float* Wph = (const float*)d_in[7];
    const float* bp  = (const float*)d_in[8];
    float* out = (float*)d_out;

    // ---- workspace layout (all sizes 256B-aligned) ----
    char* ws = (char*)d_ws;
    size_t off = 0;
    auto take = [&](size_t bytes) { char* p = ws + off; off += (bytes + 255) & ~size_t(255); return p; };

    unsigned short* W1T  = (unsigned short*)take((size_t)N4H * K1 * 2);     //  2.5 MB
    unsigned short* W2T  = (unsigned short*)take((size_t)N4H * K2 * 2);     //  4   MB
    unsigned short* xbf  = (unsigned short*)take((size_t)B_ * T_ * I_ * 2); // 16   MB
    unsigned short* h1b  = (unsigned short*)take((size_t)B_ * H_ * 2);
    unsigned short* h2b  = (unsigned short*)take((size_t)B_ * H_ * 2);
    float* c1  = (float*)take((size_t)B_ * H_ * 4);
    float* c2  = (float*)take((size_t)B_ * H_ * 4);
    float* h1f = (float*)take((size_t)B_ * H_ * 4);
    float* h2f = (float*)take((size_t)B_ * H_ * 4);
    float* z1  = (float*)take((size_t)B_ * N4H * 4);                        //  2   MB
    float* z2  = (float*)take((size_t)B_ * N4H * 4);                        //  2   MB

    // ---- one-time prep ----
    {
        int tot = N4H * K1 + N4H * K2;
        prep_weights<<<(tot + 255) / 256, 256, 0, stream>>>(Wx1, Wh1, Wx2, Wh2, W1T, W2T);
        int nx = B_ * T_ * I_;
        convert_x<<<(nx + 255) / 256, 256, 0, stream>>>(x, xbf, nx);
        zero_state<<<(B_ * H_) / 256, 256, 0, stream>>>(h1b, h2b, c1, c2, h1f, h2f);
    }

    dim3 ggrid(B_ / 128, N4H / 64);   // 2 x 32 workgroups
    dim3 gblk(256);
    int ublk = (B_ * H_) / 256;       // 512 blocks

    // ---- sequential scan over time ----
    for (int t = 0; t < T_; ++t) {
        // layer 1: A = [x_t (K=128) | h1 (K=512)],  B = W1T [2048][640]
        lstm_gemm_wmma<<<ggrid, gblk, 0, stream>>>(
            xbf + (size_t)t * I_, T_ * I_,   // row b at (b*T + t)*I
            h1b, H_,
            I_, K1, W1T, z1);
        lstm_update<<<ublk, 256, 0, stream>>>(z1, b1, c1, h1b, h1f);

        // layer 2: A = [h1 (K=512) | h2 (K=512)],  B = W2T [2048][1024]
        lstm_gemm_wmma<<<ggrid, gblk, 0, stream>>>(
            h1b, H_,
            h2b, H_,
            H_, K2, W2T, z2);
        lstm_update<<<ublk, 256, 0, stream>>>(z2, b2, c2, h2b, h2f);
    }

    // ---- head ----
    proj_kernel<<<(B_ * C_ + 255) / 256, 256, 0, stream>>>(h2f, Wph, bp, out);
}
